// MiniLLM_31963146617130
// MI455X (gfx1250) — matrix operand order repaired and checked by reading, not compiled, not run on hardware
//
#include <hip/hip_runtime.h>
#include <hip/hip_bf16.h>
#include <math.h>

// ---------------------------------------------------------------------------
// MiniLLM forward for MI455X (gfx1250): bf16 WMMA everywhere, fp32 accum.
// ---------------------------------------------------------------------------

#define S_LEN  2048
#define DMODEL 768
#define NHEAD  12
#define HDIM   64
#define DFF_N  3072
#define VOCAB  32000

typedef __attribute__((ext_vector_type(16))) __bf16 v16bf;
typedef __attribute__((ext_vector_type(8)))  float  v8f;
typedef __attribute__((ext_vector_type(4)))  int    v4i;

union BF16Frag { v16bf v; unsigned u[8]; uint4 q[2]; };

#if __has_builtin(__builtin_amdgcn_global_load_async_to_lds_b128)
#define HAVE_ASYNC_LDS 1
typedef __attribute__((address_space(1))) v4i gv4i;
typedef __attribute__((address_space(3))) v4i lv4i;
#else
#define HAVE_ASYNC_LDS 0
#endif

__device__ __forceinline__ unsigned short f2bf(float f) {
  unsigned u = __float_as_uint(f);
  unsigned r = u + 0x7fffu + ((u >> 16) & 1u);   // round-to-nearest-even
  return (unsigned short)(r >> 16);
}
__device__ __forceinline__ unsigned packbf(float lo, float hi) {
  return (unsigned)f2bf(lo) | ((unsigned)f2bf(hi) << 16);
}
__device__ __forceinline__ v8f vz8() {
  v8f z = {0.f,0.f,0.f,0.f,0.f,0.f,0.f,0.f};
  return z;
}
// A-fragment K index for (vgpr v, lane-half hi): 16-bit A 16x32 layout
__device__ __forceinline__ int akk(int v, int hi) {
  return (v < 4 ? 2 * v : 16 + 2 * (v - 4)) + 8 * hi;
}

// ---------------------------------------------------------------------------
// fp32 -> bf16 conversion (weights)
// ---------------------------------------------------------------------------
__global__ void cvt_kernel(const float* __restrict__ in,
                           unsigned short* __restrict__ out, size_t n) {
  size_t i = (size_t)blockIdx.x * blockDim.x + threadIdx.x;
  size_t stride = (size_t)gridDim.x * blockDim.x;
  for (; i < n; i += stride) out[i] = f2bf(in[i]);
}

// ---------------------------------------------------------------------------
// embedding: x = embed_w[id]*sqrt(D) + pos_embed  (fp32 + bf16 copies)
// ---------------------------------------------------------------------------
__global__ __launch_bounds__(256) void embed_kernel(
    const int* __restrict__ ids, const float* __restrict__ ew,
    const float* __restrict__ pe, float* __restrict__ x,
    unsigned short* __restrict__ xb) {
  int s = blockIdx.x, tid = threadIdx.x;
  int id = ids[s];
  size_t base = (size_t)s * DMODEL;
#pragma unroll
  for (int i = 0; i < 3; ++i) {
    int d = tid + i * 256;
    float v = ew[(size_t)id * DMODEL + d] * 27.712812921102035f + pe[base + d];
    x[base + d] = v;
    xb[base + d] = f2bf(v);
  }
}

// ---------------------------------------------------------------------------
// residual-add + LayerNorm: y = LN(a + r) * g + b   (fp32 optional + bf16)
// ---------------------------------------------------------------------------
__global__ __launch_bounds__(256) void ln_kernel(
    const float* __restrict__ a, const float* __restrict__ r,
    const float* __restrict__ g, const float* __restrict__ be,
    float* __restrict__ y, unsigned short* __restrict__ yb) {
  int row = blockIdx.x, tid = threadIdx.x;
  size_t base = (size_t)row * DMODEL;
  float v[3];
  float s1 = 0.f, s2 = 0.f;
#pragma unroll
  for (int i = 0; i < 3; ++i) {
    int d = tid + i * 256;
    float xv = a[base + d] + r[base + d];
    v[i] = xv; s1 += xv; s2 += xv * xv;
  }
#pragma unroll
  for (int off = 16; off > 0; off >>= 1) {
    s1 += __shfl_xor(s1, off);
    s2 += __shfl_xor(s2, off);
  }
  __shared__ float w1s[8], w2s[8];
  int wid = tid >> 5, lane = tid & 31;
  if (lane == 0) { w1s[wid] = s1; w2s[wid] = s2; }
  __syncthreads();
  s1 = 0.f; s2 = 0.f;
#pragma unroll
  for (int i = 0; i < 8; ++i) { s1 += w1s[i]; s2 += w2s[i]; }
  float mu  = s1 * (1.0f / DMODEL);
  float var = s2 * (1.0f / DMODEL) - mu * mu;
  float inv = rsqrtf(var + 1e-5f);
#pragma unroll
  for (int i = 0; i < 3; ++i) {
    int d = tid + i * 256;
    float o = (v[i] - mu) * inv * g[d] + be[d];
    if (y) y[base + d] = o;
    yb[base + d] = f2bf(o);
  }
}

// ---------------------------------------------------------------------------
// GEMM: C[M,N] = A[M,K] @ W[N,K]^T + bias, optional exact GELU.
// 256 threads = 8 waves (4x2), block tile 128x128, wave tile 32x64,
// K-step 32 == one v_wmma_f32_16x16x32_bf16 per fragment pair.
// ---------------------------------------------------------------------------
#define LDA_S 40    // 80B rows: 16B aligned for b128, conflict-spread
#define LDB_S 136   // sizes Bs only; Bs is used as [128][34] (68B rows)

template <int WF32, int WBF16, int GELU, int BIAS>
__global__ __launch_bounds__(256) void gemm_kernel(
    const unsigned short* __restrict__ A, const unsigned short* __restrict__ W,
    const float* __restrict__ bias, float* __restrict__ C,
    unsigned short* __restrict__ Cb, int M, int N, int K) {
  __shared__ unsigned short As[128 * LDA_S];
  __shared__ unsigned short Bs[32 * LDB_S];
  int tid = threadIdx.x;
  int wid = tid >> 5, lane = tid & 31;
  int wm = wid & 3, wn = wid >> 2;
  int m15 = lane & 15, hi = lane >> 4;
  int tileM = blockIdx.y * 128, tileN = blockIdx.x * 128;

  v8f acc[2][4];
#pragma unroll
  for (int i = 0; i < 2; ++i)
#pragma unroll
    for (int j = 0; j < 4; ++j) acc[i][j] = vz8();

  int row = tid >> 1, seg = tid & 1;
  const unsigned short* ga = A + (size_t)(tileM + row) * K + seg * 16;
  const unsigned short* gw = W + (size_t)(tileN + row) * K + seg * 16;
  unsigned short* la = &As[row * LDA_S + seg * 16];

  for (int k0 = 0; k0 < K; k0 += 32) {
    // ---- stage A tile (128 x 32 bf16), async->LDS when toolchain supports it
#if HAVE_ASYNC_LDS
    {
      __builtin_amdgcn_global_load_async_to_lds_b128(
          (gv4i*)(size_t)(ga + k0), (lv4i*)(void*)la, 0, 0);
      __builtin_amdgcn_global_load_async_to_lds_b128(
          (gv4i*)(size_t)(ga + k0 + 8), (lv4i*)(void*)(la + 8), 0, 0);
    }
#else
    {
      const uint4* s4 = (const uint4*)(ga + k0);
      uint4 d0 = s4[0], d1 = s4[1];
      uint4* dA = (uint4*)la;
      dA[0] = d0; dA[1] = d1;
    }
#endif
    // ---- stage W tile into Bs[n][k] (128 x 32 bf16, row stride 34)
    {
      const uint4* s4 = (const uint4*)(gw + k0);
      uint4 d0 = s4[0], d1 = s4[1];
      unsigned sv[8] = {d0.x, d0.y, d0.z, d0.w, d1.x, d1.y, d1.z, d1.w};
#pragma unroll
      for (int i = 0; i < 8; ++i) {
        *(unsigned*)&Bs[row * 34 + seg * 16 + 2 * i] = sv[i];
      }
    }
    if (k0 + 32 < K) {
      __builtin_prefetch((const void*)(ga + k0 + 32), 0, 1);
      __builtin_prefetch((const void*)(gw + k0 + 32), 0, 1);
    }
#if HAVE_ASYNC_LDS
#if __has_builtin(__builtin_amdgcn_s_wait_asynccnt)
    __builtin_amdgcn_s_wait_asynccnt(0);
#else
    asm volatile("s_wait_asynccnt 0" ::: "memory");
#endif
#endif
    __syncthreads();

    // ---- fragments
    BF16Frag af[2];
#pragma unroll
    for (int fm = 0; fm < 2; ++fm) {
      int mr = wm * 32 + fm * 16 + m15;
#pragma unroll
      for (int v = 0; v < 8; ++v)
        af[fm].u[v] = *(const unsigned*)&As[mr * LDA_S + akk(v, hi)];
    }
    BF16Frag bfr[4];
#pragma unroll
    for (int fn = 0; fn < 4; ++fn) {
#pragma unroll
      for (int v = 0; v < 8; ++v)
        bfr[fn].u[v] = *(const unsigned*)&Bs[(wn * 64 + fn * 16 + m15) * 34 + akk(v, hi)];
    }
#pragma unroll
    for (int fm = 0; fm < 2; ++fm)
#pragma unroll
      for (int fn = 0; fn < 4; ++fn)
        acc[fm][fn] = __builtin_amdgcn_wmma_f32_16x16x32_bf16(
            false, af[fm].v, false, bfr[fn].v, (short)0, acc[fm][fn], false, false);
    __syncthreads();
  }

  // ---- epilogue: C element (M = v+8*hi, N = lane&15) per frag
#pragma unroll
  for (int fn = 0; fn < 4; ++fn) {
    int col = tileN + wn * 64 + fn * 16 + m15;
    float bv = BIAS ? bias[col] : 0.f;
#pragma unroll
    for (int fm = 0; fm < 2; ++fm) {
#pragma unroll
      for (int v = 0; v < 8; ++v) {
        int rrow = tileM + wm * 32 + fm * 16 + v + 8 * hi;
        float xv = acc[fm][fn][v] + bv;
        if (GELU) xv = 0.5f * xv * (1.f + erff(xv * 0.70710678118f));
        size_t idx = (size_t)rrow * N + col;
        if (WF32)  C[idx]  = xv;
        if (WBF16) Cb[idx] = f2bf(xv);
      }
    }
  }
}

// ---------------------------------------------------------------------------
// Flash attention: block = (head, 128 queries), 8 waves x 16 query rows.
// Per 32-key block: scores via 2 WMMA (K over HD=64), online softmax with
// cross-lane row reductions, P bounced through wave-private LDS to convert
// C-layout -> A-layout, then P@V via WMMA.
// ---------------------------------------------------------------------------
#define KT_S 40   // sizes Kt; used as Kt[32][72] (K block as [key][d], 16B-aligned rows)
#define VS_S 72   // sizes Vs; used as Vs[64][34] (V transposed, [d][key])
#define PS_S 34   // Ps per wave [16][34]

__global__ __launch_bounds__(256) void attn_kernel(
    const float* __restrict__ qkv, unsigned short* __restrict__ ob) {
  __shared__ unsigned short Kt[64 * KT_S];
  __shared__ unsigned short Vs[32 * VS_S];
  __shared__ unsigned short Ps[8][16 * PS_S];
  int head = blockIdx.y;
  int q0 = blockIdx.x * 128;
  int tid = threadIdx.x, wid = tid >> 5, lane = tid & 31;
  int m15 = lane & 15, hi = lane >> 4;
  int qrow0 = q0 + wid * 16;

  // Q fragments (16 rows x 64 hd -> 2 A-frags), loaded straight from qkv
  BF16Frag qf[2];
#pragma unroll
  for (int f = 0; f < 2; ++f) {
    const float* qp =
        qkv + (size_t)(qrow0 + m15) * (3 * DMODEL) + head * HDIM + f * 32;
#pragma unroll
    for (int v = 0; v < 8; ++v) {
      int kk = akk(v, hi);
      qf[f].u[v] = packbf(qp[kk], qp[kk + 1]);
    }
  }

  v8f oacc[4];
#pragma unroll
  for (int i = 0; i < 4; ++i) oacc[i] = vz8();
  float m_st[8], l_st[8];
#pragma unroll
  for (int v = 0; v < 8; ++v) { m_st[v] = -3.0e38f; l_st[v] = 0.f; }

  int key_r = tid >> 3;        // 0..31
  int dseg  = (tid & 7) * 8;   // 0..56

  for (int kb = 0; kb < q0 + 128; kb += 32) {
    // ---- stage K (Kt[key][d], stride 72) and V (transposed to Vs[d][key], stride 34) as bf16
    {
      const float* kp = qkv + (size_t)(kb + key_r) * (3 * DMODEL) + DMODEL +
                        head * HDIM + dseg;
      const float* vp = kp + DMODEL;
#pragma unroll
      for (int i = 0; i < 8; ++i) Kt[key_r * 72 + dseg + i] = f2bf(kp[i]);
#pragma unroll
      for (int i = 0; i < 8; ++i) Vs[(dseg + i) * 34 + key_r] = f2bf(vp[i]);
    }
    __syncthreads();

    if (kb <= qrow0 + 15) {   // wave-uniform: EXEC all-ones at WMMA sites
      // ---- scores: 16 rows x 32 keys = 2 C-frags, accumulate over HD halves
      v8f sc[2];
#pragma unroll
      for (int j = 0; j < 2; ++j) {
        v8f s = vz8();
#pragma unroll
        for (int f = 0; f < 2; ++f) {
          BF16Frag bb;
#pragma unroll
          for (int v = 0; v < 8; ++v)
            bb.u[v] = *(const unsigned*)&Kt[(j * 16 + m15) * 72 + f * 32 + akk(v, hi)];
          s = __builtin_amdgcn_wmma_f32_16x16x32_bf16(
              false, qf[f].v, false, bb.v, (short)0, s, false, false);
        }
#pragma unroll
        for (int v = 0; v < 8; ++v) {
          int r = qrow0 + v + 8 * hi;
          int c = kb + j * 16 + m15;
          s[v] = (c <= r) ? s[v] * 0.125f : -3.0e38f;
        }
        sc[j] = s;
      }
      // ---- online softmax (row stats reduced across the 16-lane half)
      float bm[8], alpha[8], rs[8];
#pragma unroll
      for (int v = 0; v < 8; ++v) bm[v] = fmaxf(sc[0][v], sc[1][v]);
#pragma unroll
      for (int off = 1; off < 16; off <<= 1)
#pragma unroll
        for (int v = 0; v < 8; ++v) bm[v] = fmaxf(bm[v], __shfl_xor(bm[v], off));
#pragma unroll
      for (int v = 0; v < 8; ++v) {
        float mn = fmaxf(m_st[v], bm[v]);
        alpha[v] = __expf(m_st[v] - mn);
        m_st[v] = mn;
        rs[v] = 0.f;
      }
#pragma unroll
      for (int j = 0; j < 2; ++j)
#pragma unroll
        for (int v = 0; v < 8; ++v) {
          float p = __expf(sc[j][v] - m_st[v]);
          rs[v] += p;
          Ps[wid][(v + 8 * hi) * PS_S + j * 16 + m15] = f2bf(p);
        }
#pragma unroll
      for (int off = 1; off < 16; off <<= 1)
#pragma unroll
        for (int v = 0; v < 8; ++v) rs[v] += __shfl_xor(rs[v], off);
#pragma unroll
      for (int v = 0; v < 8; ++v) l_st[v] = l_st[v] * alpha[v] + rs[v];
#pragma unroll
      for (int nf = 0; nf < 4; ++nf)
#pragma unroll
        for (int v = 0; v < 8; ++v) oacc[nf][v] *= alpha[v];

      // ---- P@V: re-read P in A-layout (same-wave LDS ops are in-order)
      BF16Frag pf;
#pragma unroll
      for (int v = 0; v < 8; ++v)
        pf.u[v] = *(const unsigned*)&Ps[wid][m15 * PS_S + akk(v, hi)];
#pragma unroll
      for (int nf = 0; nf < 4; ++nf) {
        BF16Frag vb;
#pragma unroll
        for (int v = 0; v < 8; ++v)
          vb.u[v] = *(const unsigned*)&Vs[(nf * 16 + m15) * 34 + akk(v, hi)];
        oacc[nf] = __builtin_amdgcn_wmma_f32_16x16x32_bf16(
            false, pf.v, false, vb.v, (short)0, oacc[nf], false, false);
      }
    }
    __syncthreads();
  }
  // ---- write o (bf16) normalized by row sums
#pragma unroll
  for (int nf = 0; nf < 4; ++nf)
#pragma unroll
    for (int v = 0; v < 8; ++v) {
      int r = qrow0 + v + 8 * hi;
      int d = nf * 16 + m15;
      ob[(size_t)r * DMODEL + head * HDIM + d] = f2bf(oacc[nf][v] / l_st[v]);
    }
}

// ---------------------------------------------------------------------------
// launch
// ---------------------------------------------------------------------------
extern "C" void kernel_launch(void* const* d_in, const int* in_sizes, int n_in,
                              void* d_out, int out_size, void* d_ws,
                              size_t ws_size, hipStream_t stream) {
  (void)in_sizes; (void)n_in; (void)out_size; (void)ws_size;
  const int*   ids  = (const int*)d_in[0];
  const float* ew   = (const float*)d_in[1];
  const float* pe   = (const float*)d_in[2];
  const float* inw  = (const float*)d_in[3];
  const float* inb  = (const float*)d_in[4];
  const float* outw = (const float*)d_in[5];
  const float* outb = (const float*)d_in[6];
  const float* ng   = (const float*)d_in[7];
  const float* nb   = (const float*)d_in[8];
  const float* w1   = (const float*)d_in[9];
  const float* b1   = (const float*)d_in[10];
  const float* w2   = (const float*)d_in[11];
  const float* b2   = (const float*)d_in[12];
  const float* lmw  = (const float*)d_in[13];
  float* logits = (float*)d_out;

  char* ws = (char*)d_ws;
  size_t off = 0;
  auto alloc = [&](size_t bytes) -> void* {
    void* p = ws + off;
    off = (off + bytes + 255) & ~(size_t)255;
    return p;
  };
  const size_t S = S_LEN, D = DMODEL, DF = DFF_N, V = VOCAB, TD = 3 * DMODEL;
  unsigned short* inwb  = (unsigned short*)alloc(TD * D * 2);
  unsigned short* outwb = (unsigned short*)alloc(D * D * 2);
  unsigned short* w1b   = (unsigned short*)alloc(DF * D * 2);
  unsigned short* w2b   = (unsigned short*)alloc(D * DF * 2);
  unsigned short* lmwb  = (unsigned short*)alloc(V * D * 2);
  float*          x     = (float*)alloc(S * D * 4);
  unsigned short* xb    = (unsigned short*)alloc(S * D * 2);
  float*          qkv   = (float*)alloc(S * TD * 4);
  unsigned short* obuf  = (unsigned short*)alloc(S * D * 2);
  float*          proj  = (float*)alloc(S * D * 4);
  float*          x1    = (float*)alloc(S * D * 4);
  unsigned short* x1b   = (unsigned short*)alloc(S * D * 2);
  unsigned short* hb    = (unsigned short*)alloc(S * DF * 2);
  float*          ffo   = (float*)alloc(S * D * 4);
  unsigned short* x2b   = (unsigned short*)alloc(S * D * 2);

  // weights fp32 -> bf16 (one streaming pass)
  cvt_kernel<<<2048, 256, 0, stream>>>(inw,  inwb,  TD * D);
  cvt_kernel<<<2048, 256, 0, stream>>>(outw, outwb, D * D);
  cvt_kernel<<<2048, 256, 0, stream>>>(w1,   w1b,   DF * D);
  cvt_kernel<<<2048, 256, 0, stream>>>(w2,   w2b,   D * DF);
  cvt_kernel<<<4096, 256, 0, stream>>>(lmw,  lmwb,  V * D);

  embed_kernel<<<2048, 256, 0, stream>>>(ids, ew, pe, x, xb);

  // QKV projection
  gemm_kernel<1, 0, 0, 1><<<dim3(TD / 128, S / 128), 256, 0, stream>>>(
      xb, inwb, inb, qkv, nullptr, (int)S, (int)TD, (int)D);
  // attention
  attn_kernel<<<dim3(S / 128, NHEAD), 256, 0, stream>>>(qkv, obuf);
  // output projection
  gemm_kernel<1, 0, 0, 1><<<dim3(D / 128, S / 128), 256, 0, stream>>>(
      obuf, outwb, outb, proj, nullptr, (int)S, (int)D, (int)D);
  // LN1
  ln_kernel<<<2048, 256, 0, stream>>>(x, proj, ng, nb, x1, x1b);
  // FFN up + GELU
  gemm_kernel<0, 1, 1, 1><<<dim3(DF / 128, S / 128), 256, 0, stream>>>(
      x1b, w1b, b1, nullptr, hb, (int)S, (int)DF, (int)D);
  // FFN down
  gemm_kernel<1, 0, 0, 1><<<dim3(D / 128, S / 128), 256, 0, stream>>>(
      hb, w2b, b2, ffo, nullptr, (int)S, (int)D, (int)DF);
  // LN2
  ln_kernel<<<2048, 256, 0, stream>>>(x1, ffo, ng, nb, nullptr, x2b);
  // LM head
  gemm_kernel<1, 0, 0, 0><<<dim3(V / 128, S / 128), 256, 0, stream>>>(
      x2b, lmwb, nullptr, logits, nullptr, (int)S, (int)V, (int)D);
}
